// InstructBlipQFormerLayerWithMoE_85667417686125
// MI455X (gfx1250) — compile-verified, matrix operand order checked by reading, not run on hardware
//
#include <hip/hip_runtime.h>
#include <hip/hip_bf16.h>

// Problem constants (from reference)
#define B_   16
#define S_   2048
#define H_   768
#define I_   3072
#define E_   8
#define T_   (B_ * S_)          // 32768 tokens

// Expert-kernel tiling
#define M_TILE   64             // tokens per block
#define ICHUNK   64             // intermediate (I) chunk
#define NWAVES   16             // 512 threads, wave32
#define BLOCK_T  (NWAVES * 32)
#define TILES_PER_E ((T_ + M_TILE - 1) / M_TILE)   // 512
#define K1_STAGE 128            // w1 K-depth staged per barrier pair

// LDS strides (elements); every fragment base/offset stays 16B aligned
#define XS_STRIDE   776         // 64 x 776 bf16  =  99328 B
#define W1S_STRIDE  136         // 64 x 136 bf16  =  17408 B  (128 K + pad)
#define HMS_STRIDE  72          // 64 x 72  bf16  =   9216 B
#define W2S_STRIDE  72          // 768 x 72 bf16  = 110592 B  (64 K + pad)

#define SMEM_XS    0
#define SMEM_W1S   (SMEM_XS  + M_TILE * XS_STRIDE * 2)
#define SMEM_HMS   (SMEM_W1S + M_TILE * W1S_STRIDE * 2)
#define SMEM_W2S   (SMEM_HMS + M_TILE * HMS_STRIDE * 2)
#define SMEM_TOK   (SMEM_W2S + H_ * W2S_STRIDE * 2)
#define SMEM_WT    (SMEM_TOK + M_TILE * 4)
#define SMEM_BYTES (SMEM_WT  + M_TILE * 4)          // ~237 KB

// CDNA5 async Global->LDS copy (ASYNCcnt), if the toolchain exposes it.
#if defined(__HIP_DEVICE_COMPILE__) && \
    __has_builtin(__builtin_amdgcn_global_load_async_to_lds_b128) && \
    __has_builtin(__builtin_amdgcn_s_wait_asynccnt)
#define ASYNC_COPY 1
#define AS1 __attribute__((address_space(1)))
#define AS3 __attribute__((address_space(3)))
typedef int g4i __attribute__((vector_size(16)));   // matches builtin pointee
#else
#define ASYNC_COPY 0
#endif

typedef unsigned short u16;
typedef __attribute__((ext_vector_type(16))) __bf16 v16bf;
typedef __attribute__((ext_vector_type(8)))  __bf16 v8bf;
typedef __attribute__((ext_vector_type(8)))  float  v8f;

__device__ __forceinline__ u16 f2bf(float f) {
    union { float f; unsigned u; } c; c.f = f;
    unsigned u = c.u;
    u += 0x7FFFu + ((u >> 16) & 1u);    // round-to-nearest-even
    return (u16)(u >> 16);
}

__device__ __forceinline__ unsigned pack2bf(float a, float b) {
    return (unsigned)f2bf(a) | ((unsigned)f2bf(b) << 16);
}

// 16B Global->LDS copy: async DMA if available, else reg round-trip.
__device__ __forceinline__ void copy16_g2l(const u16* src, u16* dst) {
#if ASYNC_COPY
    __builtin_amdgcn_global_load_async_to_lds_b128(
        (AS1 g4i*)src, (AS3 g4i*)dst, 0, 0);
#else
    *reinterpret_cast<uint4*>(dst) = *reinterpret_cast<const uint4*>(src);
#endif
}

__device__ __forceinline__ void copy_fence() {
#if ASYNC_COPY
    __builtin_amdgcn_s_wait_asynccnt(0);
#endif
}

// Branchless GeLU (tanh form) on native v_exp_f32 / v_rcp_f32.
__device__ __forceinline__ float gelu_fast(float u) {
    const float z  = 0.7978845608028654f * (u + 0.044715f * u * u * u);
    const float ez = __expf(2.0f * z);
    const float th = 1.0f - 2.0f * __builtin_amdgcn_rcpf(ez + 1.0f);
    return 0.5f * u * (1.0f + th);
}

union FragU { v16bf v; v8bf h[2]; };

// Load a 16x32 bf16 A-style fragment (or B fragment stored [n][k]) from LDS.
// ISA layout: lanes 0-15 -> row=lane, K {k0+0..7, k0+16..23};
//             lanes 16-31 -> row=lane-16, K {k0+8..15, k0+24..31}.
__device__ __forceinline__ v16bf ld_frag(const u16* base, int stride, int k0) {
    const int lane = threadIdx.x & 31;
    const int r    = lane & 15;
    const int sel  = lane >> 4;
    const u16* p = base + r * stride + k0 + sel * 8;
    FragU u;
    u.h[0] = *reinterpret_cast<const v8bf*>(p);        // ds_load_b128
    u.h[1] = *reinterpret_cast<const v8bf*>(p + 16);   // ds_load_b128
    return u.v;
}

__device__ __forceinline__ v8f vzero8() {
    v8f v;
#pragma unroll
    for (int i = 0; i < 8; ++i) v[i] = 0.0f;
    return v;
}

// ---------------------------------------------------------------------------
// Kernel 0: tiled transpose + fp32->bf16 convert:  in [E][M][N] -> out [E][N][M].
// ---------------------------------------------------------------------------
__global__ __launch_bounds__(256) void transpose_bf16_kernel(
    const float* __restrict__ in, u16* __restrict__ outp, int M, int N)
{
    __shared__ u16 tile[64][72];
    const int e  = blockIdx.z;
    const int r0 = blockIdx.y * 64;
    const int c0 = blockIdx.x * 64;
    const int tid = threadIdx.x;
    const float* src = in   + (size_t)e * M * N;
    u16*         dst = outp + (size_t)e * M * N;

    const int rr = tid >> 4;         // 0..15
    const int cc = (tid & 15) * 4;   // 0..60
#pragma unroll
    for (int rb = 0; rb < 64; rb += 16) {
        const float4 f = *reinterpret_cast<const float4*>(
            src + (size_t)(r0 + rb + rr) * N + c0 + cc);
        tile[cc + 0][rb + rr] = f2bf(f.x);
        tile[cc + 1][rb + rr] = f2bf(f.y);
        tile[cc + 2][rb + rr] = f2bf(f.z);
        tile[cc + 3][rb + rr] = f2bf(f.w);
    }
    __syncthreads();
#pragma unroll
    for (int q = tid; q < 512; q += 256) {
        const int row = q >> 3;          // 0..63 : output row (= input col)
        const int seg = (q & 7) * 8;     // 0..56 : output col (= input row)
        const uint4 v = *reinterpret_cast<const uint4*>(&tile[row][seg]);
        *reinterpret_cast<uint4*>(dst + (size_t)(c0 + row) * M + r0 + seg) = v;
    }
}

// ---------------------------------------------------------------------------
// Kernel 1: router. One wave per token.
// ---------------------------------------------------------------------------
__global__ __launch_bounds__(256) void moe_router_kernel(
    const float* __restrict__ x,        // [T, H]
    const float* __restrict__ gate_w,   // [E, H]
    float* __restrict__ logits_out,     // [T, E]
    int*   __restrict__ cnt,            // [E]
    int*   __restrict__ tok_list,       // [E, T]
    float* __restrict__ w_list)         // [E, T]
{
    __shared__ float gws[E_ * H_];      // 24 KB
    for (int i = threadIdx.x; i < E_ * H_; i += 256) gws[i] = gate_w[i];
    __syncthreads();

    const int wave = threadIdx.x >> 5;
    const int lane = threadIdx.x & 31;
    const int t = blockIdx.x * 8 + wave;
    if (t >= T_) return;

    const float* xr = x + (size_t)t * H_;
    float acc[E_];
#pragma unroll
    for (int e = 0; e < E_; ++e) acc[e] = 0.0f;

    for (int h = lane; h < H_; h += 32) {
        const float xv = xr[h];
#pragma unroll
        for (int e = 0; e < E_; ++e) acc[e] += xv * gws[e * H_ + h];
    }
#pragma unroll
    for (int off = 16; off > 0; off >>= 1) {
#pragma unroll
        for (int e = 0; e < E_; ++e) acc[e] += __shfl_xor(acc[e], off, 32);
    }

    if (lane == 0) {
        float mx = acc[0];
#pragma unroll
        for (int e = 1; e < E_; ++e) mx = fmaxf(mx, acc[e]);
        float p[E_];
#pragma unroll
        for (int e = 0; e < E_; ++e) p[e] = __expf(acc[e] - mx);

        int i0 = 0;
#pragma unroll
        for (int e = 1; e < E_; ++e) if (p[e] > p[i0]) i0 = e;
        int i1 = (i0 == 0) ? 1 : 0;
#pragma unroll
        for (int e = 0; e < E_; ++e) if (e != i0 && p[e] > p[i1]) i1 = e;

        const float inv = 1.0f / (p[i0] + p[i1]);
#pragma unroll
        for (int e = 0; e < E_; ++e) logits_out[(size_t)t * E_ + e] = acc[e];

        int pos0 = atomicAdd(&cnt[i0], 1);
        tok_list[i0 * T_ + pos0] = t;
        w_list [i0 * T_ + pos0] = p[i0] * inv;
        int pos1 = atomicAdd(&cnt[i1], 1);
        tok_list[i1 * T_ + pos1] = t;
        w_list [i1 * T_ + pos1] = p[i1] * inv;
    }
}

// ---------------------------------------------------------------------------
// Kernel 2: fused expert FFN. Block = 512 thr (16 waves), 64 gathered tokens.
// PRE=true: weights pre-converted/pre-transposed bf16 in workspace; staging is
// async Global->LDS DMA. PRE=false: convert fp32->bf16 while staging.
// ---------------------------------------------------------------------------
template <bool PRE>
__global__ __launch_bounds__(BLOCK_T) void moe_expert_kernel(
    const float* __restrict__ x,        // [T, H]
    const float* __restrict__ w1,       // [E, H, I]
    const float* __restrict__ pb1,      // [E, I]
    const float* __restrict__ w2,       // [E, I, H]
    const float* __restrict__ pb2,      // [E, H]
    const u16*   __restrict__ w1t,      // [E, I, H] bf16 (PRE)
    const u16*   __restrict__ w2t,      // [E, H, I] bf16 (PRE)
    float*       __restrict__ out,      // [T, H], pre-zeroed
    const int*   __restrict__ cnt,
    const int*   __restrict__ tok_list,
    const float* __restrict__ w_list)
{
    extern __shared__ char smem[];
    u16*      xs    = reinterpret_cast<u16*>(smem + SMEM_XS);
    u16*      w1s   = reinterpret_cast<u16*>(smem + SMEM_W1S);
    unsigned* w1s32 = reinterpret_cast<unsigned*>(smem + SMEM_W1S);
    u16*      hms   = reinterpret_cast<u16*>(smem + SMEM_HMS);
    u16*      w2s   = reinterpret_cast<u16*>(smem + SMEM_W2S);
    unsigned* w2s32 = reinterpret_cast<unsigned*>(smem + SMEM_W2S);
    int*      toks  = reinterpret_cast<int*>(smem + SMEM_TOK);
    float*    wts   = reinterpret_cast<float*>(smem + SMEM_WT);

    const int e  = blockIdx.y;
    const int ne = cnt[e];
    const int m0 = blockIdx.x * M_TILE;
    if (m0 >= ne) return;
    const int rows = min(M_TILE, ne - m0);

    const int tid  = threadIdx.x;
    const int wave = tid >> 5;
    const int lane = tid & 31;
    const int wm   = wave >> 2;   // 0..3 : 16-row group
    const int wn   = wave & 3;    // 0..3 : column group

    for (int i = tid; i < M_TILE; i += BLOCK_T) {
        toks[i] = (i < rows) ? tok_list[e * T_ + m0 + i] : -1;
        wts[i]  = (i < rows) ? w_list [e * T_ + m0 + i] : 0.0f;
    }
    __syncthreads();

    // Gather token rows, fp32 -> bf16, into LDS x slab.
    for (int r = wave; r < M_TILE; r += NWAVES) {
        const int tk = toks[r];
        const float4* src = (tk >= 0)
            ? reinterpret_cast<const float4*>(x + (size_t)tk * H_) : nullptr;
        for (int c4 = lane; c4 < H_ / 4; c4 += 32) {
            float4 f = src ? src[c4] : make_float4(0.f, 0.f, 0.f, 0.f);
            uint2* dst = reinterpret_cast<uint2*>(xs + r * XS_STRIDE + c4 * 4);
            *dst = make_uint2(pack2bf(f.x, f.y), pack2bf(f.z, f.w));
        }
    }

    v8f yacc[12];
#pragma unroll
    for (int i = 0; i < 12; ++i) yacc[i] = vzero8();

#pragma unroll 1
    for (int chunk = 0; chunk < I_ / ICHUNK; ++chunk) {
        const int i0 = chunk * ICHUNK;

        // ---- GEMM1: hmid[64,64] = x[64,768] @ w1[768, i0:i0+64] ----
        v8f cacc = vzero8();
#pragma unroll 1
        for (int ks = 0; ks < H_ / K1_STAGE; ++ks) {        // 6 stages
            __syncthreads();
            if (PRE) {
                // w1t rows are [i][k]; slab row = 256 B. 1024 x 16B chunks.
#pragma unroll
                for (int it = 0; it < 2; ++it) {
                    const int q   = tid + it * BLOCK_T;
                    const int i   = q >> 4;            // 0..63
                    const int seg = (q & 15) * 8;      // 0..120 (k elements)
                    copy16_g2l(
                        w1t + ((size_t)e * I_ + i0 + i) * H_ + ks * K1_STAGE + seg,
                        w1s + i * W1S_STRIDE + seg);
                }
                if (ks + 1 < H_ / K1_STAGE) {
                    const u16* pf = w1t + ((size_t)e * I_ + i0 + (tid >> 1)) * H_
                                    + (ks + 1) * K1_STAGE + (tid & 1) * 64;
                    __builtin_prefetch(pf, 0, 1);
                }
                copy_fence();
            } else {
                // Convert while staging: pack (k,k+1) bf16 pairs -> b32 stores.
#pragma unroll
                for (int q = tid; q < (K1_STAGE / 2) * (ICHUNK / 4); q += BLOCK_T) {
                    const int i4   = (q & 15) * 4;
                    const int kp   = q >> 4;
                    const int krow = ks * K1_STAGE + kp * 2;
                    const float4 f0 = *reinterpret_cast<const float4*>(
                        w1 + ((size_t)e * H_ + krow) * I_ + i0 + i4);
                    const float4 f1 = *reinterpret_cast<const float4*>(
                        w1 + ((size_t)e * H_ + krow + 1) * I_ + i0 + i4);
                    w1s32[(i4 + 0) * (W1S_STRIDE / 2) + kp] = pack2bf(f0.x, f1.x);
                    w1s32[(i4 + 1) * (W1S_STRIDE / 2) + kp] = pack2bf(f0.y, f1.y);
                    w1s32[(i4 + 2) * (W1S_STRIDE / 2) + kp] = pack2bf(f0.z, f1.z);
                    w1s32[(i4 + 3) * (W1S_STRIDE / 2) + kp] = pack2bf(f0.w, f1.w);
                }
                if (ks + 1 < H_ / K1_STAGE) {
                    const float* pf = w1 + ((size_t)e * H_ +
                        (ks + 1) * K1_STAGE + (tid >> 2)) * I_ + i0 + (tid & 3) * 16;
                    __builtin_prefetch(pf, 0, 1);
                }
            }
            __syncthreads();
#pragma unroll
            for (int kki = 0; kki < K1_STAGE / 32; ++kki) { // 4 WMMAs / stage
                const v16bf a = ld_frag(xs + (wm * 16) * XS_STRIDE, XS_STRIDE,
                                        ks * K1_STAGE + kki * 32);
                const v16bf b = ld_frag(w1s + (wn * 16) * W1S_STRIDE, W1S_STRIDE,
                                        kki * 32);
                cacc = __builtin_amdgcn_wmma_f32_16x16x32_bf16(
                           false, a, false, b, (short)0, cacc, false, false);
            }
        }

        // Bias + fast GeLU, bf16 store to hmid slab.
        {
            const int nlo  = lane & 15;
            const int half = lane >> 4;
            const float bias = pb1[(size_t)e * I_ + i0 + wn * 16 + nlo];
#pragma unroll
            for (int r = 0; r < 8; ++r) {
                const float v = gelu_fast(cacc[r] + bias);
                const int row = wm * 16 + r + 8 * half;
                hms[row * HMS_STRIDE + wn * 16 + nlo] = f2bf(v);
            }
        }

        // ---- GEMM2: y[64,768] += hmid[64,64] @ w2[i0:i0+64, 768] ----
        __syncthreads();
        if (PRE) {
            // w2t rows are [h][k]; full 64-K slab, row = 128 B. 6144 x 16B.
#pragma unroll
            for (int it = 0; it < 12; ++it) {
                const int q   = tid + it * BLOCK_T;
                const int h   = q >> 3;            // 0..767
                const int seg = (q & 7) * 8;       // 0..56 (k elements)
                copy16_g2l(w2t + ((size_t)e * H_ + h) * I_ + i0 + seg,
                           w2s + h * W2S_STRIDE + seg);
            }
            if (chunk + 1 < I_ / ICHUNK) {
                const u16* pf = w2t + ((size_t)e * H_ + tid) * I_ + i0 + ICHUNK;
                __builtin_prefetch(pf, 0, 1);
            }
            copy_fence();
        } else {
            // Convert while staging the full 64-K slab.
#pragma unroll
            for (int q = tid; q < 32 * (H_ / 4); q += BLOCK_T) {
                const int h4   = (q % (H_ / 4)) * 4;
                const int kp   = q / (H_ / 4);     // 0..31 (k-pair)
                const int irow = i0 + kp * 2;
                const float4 f0 = *reinterpret_cast<const float4*>(
                    w2 + ((size_t)e * I_ + irow) * H_ + h4);
                const float4 f1 = *reinterpret_cast<const float4*>(
                    w2 + ((size_t)e * I_ + irow + 1) * H_ + h4);
                w2s32[(h4 + 0) * (W2S_STRIDE / 2) + kp] = pack2bf(f0.x, f1.x);
                w2s32[(h4 + 1) * (W2S_STRIDE / 2) + kp] = pack2bf(f0.y, f1.y);
                w2s32[(h4 + 2) * (W2S_STRIDE / 2) + kp] = pack2bf(f0.z, f1.z);
                w2s32[(h4 + 3) * (W2S_STRIDE / 2) + kp] = pack2bf(f0.w, f1.w);
            }
            if (chunk + 1 < I_ / ICHUNK) {
                const float* pf = w2 + ((size_t)e * I_ + i0 + ICHUNK + (tid >> 4)) * H_
                                  + (tid & 15) * 16;
                __builtin_prefetch(pf, 0, 1);
            }
        }
        __syncthreads();
#pragma unroll
        for (int k2 = 0; k2 < 2; ++k2) {
            const int kbase = k2 * 32;
            const v16bf a2 = ld_frag(hms + (wm * 16) * HMS_STRIDE, HMS_STRIDE, kbase);
#pragma unroll
            for (int tn = 0; tn < 12; ++tn) {
                const v16bf b2f = ld_frag(
                    w2s + (wn * 192 + tn * 16) * W2S_STRIDE, W2S_STRIDE, kbase);
                yacc[tn] = __builtin_amdgcn_wmma_f32_16x16x32_bf16(
                               false, a2, false, b2f, (short)0, yacc[tn], false, false);
            }
        }
    }

    // Epilogue: + b2, scale by routing weight, atomic-accumulate to out.
    {
        const int nlo  = lane & 15;
        const int half = lane >> 4;
#pragma unroll
        for (int tn = 0; tn < 12; ++tn) {
            const int col = wn * 192 + tn * 16 + nlo;
            const float bias = pb2[(size_t)e * H_ + col];
#pragma unroll
            for (int r = 0; r < 8; ++r) {
                const int row = wm * 16 + r + 8 * half;
                if (row < rows) {
                    const float v = (yacc[tn][r] + bias) * wts[row];
                    unsafeAtomicAdd(out + (size_t)toks[row] * H_ + col, v);
                }
            }
        }
    }
}

// ---------------------------------------------------------------------------
// Host launcher. Inputs: hidden_states, gate_w, w1, b1, w2, b2.
// Output: [out (B,S,H) | router_logits (T,E)] as fp32.
// ---------------------------------------------------------------------------
extern "C" void kernel_launch(void* const* d_in, const int* in_sizes, int n_in,
                              void* d_out, int out_size, void* d_ws, size_t ws_size,
                              hipStream_t stream) {
    const float* x      = (const float*)d_in[0];
    const float* gate_w = (const float*)d_in[1];
    const float* w1     = (const float*)d_in[2];
    const float* b1     = (const float*)d_in[3];
    const float* w2     = (const float*)d_in[4];
    const float* b2     = (const float*)d_in[5];

    float* out    = (float*)d_out;                       // [T, H]
    float* logits = (float*)d_out + (size_t)T_ * H_;     // [T, E]

    int*   cnt      = (int*)d_ws;                                    // 64 B reserved
    int*   tok_list = (int*)((char*)d_ws + 64);                      // E*T ints
    float* w_list   = (float*)((char*)d_ws + 64 + (size_t)E_ * T_ * 4);

    const size_t disp_bytes = (64 + (size_t)E_ * T_ * 4 * 2 + 255) & ~(size_t)255;
    const size_t wconv = (size_t)E_ * H_ * I_ * sizeof(u16);   // 37.75 MB each
    const bool   pre   = ws_size >= disp_bytes + 2 * wconv;
    u16* w1t = (u16*)((char*)d_ws + disp_bytes);
    u16* w2t = w1t + (size_t)E_ * H_ * I_;

    (void)hipMemsetAsync(d_ws, 0, 64, stream);
    (void)hipMemsetAsync(out, 0, (size_t)T_ * H_ * sizeof(float), stream);

    if (pre) {
        // w1 [E][H][I] -> w1t [E][I][H];  w2 [E][I][H] -> w2t [E][H][I]
        transpose_bf16_kernel<<<dim3(I_ / 64, H_ / 64, E_), 256, 0, stream>>>(
            w1, w1t, H_, I_);
        transpose_bf16_kernel<<<dim3(H_ / 64, I_ / 64, E_), 256, 0, stream>>>(
            w2, w2t, I_, H_);
    }

    moe_router_kernel<<<T_ / 8, 256, 0, stream>>>(
        x, gate_w, logits, cnt, tok_list, w_list);

    dim3 grid(TILES_PER_E, E_);
    if (pre) {
        (void)hipFuncSetAttribute((const void*)moe_expert_kernel<true>,
                            hipFuncAttributeMaxDynamicSharedMemorySize, SMEM_BYTES);
        moe_expert_kernel<true><<<grid, BLOCK_T, SMEM_BYTES, stream>>>(
            x, w1, b1, w2, b2, w1t, w2t, out, cnt, tok_list, w_list);
    } else {
        (void)hipFuncSetAttribute((const void*)moe_expert_kernel<false>,
                            hipFuncAttributeMaxDynamicSharedMemorySize, SMEM_BYTES);
        moe_expert_kernel<false><<<grid, BLOCK_T, SMEM_BYTES, stream>>>(
            x, w1, b1, w2, b2, w1t, w2t, out, cnt, tok_list, w_list);
    }
}
